// GraphAttentionDecoder_68015102099592
// MI455X (gfx1250) — matrix operand order repaired and checked by reading, not compiled, not run on hardware
//
#include <hip/hip_runtime.h>

namespace {
constexpr int kB  = 512;   // batch
constexpr int kN  = 129;   // nodes
constexpr int kD  = 256;   // d_model
constexpr int kH  = 8;     // heads
constexpr int kHD = 32;    // head depth
constexpr int kT  = 128;   // decode steps
constexpr float kClip = 10.0f;
constexpr float kCap  = 1.0f;
}

typedef __attribute__((ext_vector_type(16))) __bf16 v16bf;
typedef __attribute__((ext_vector_type(8)))  float  v8f;

union FragU { uint4 q[2]; v16bf v; };

__device__ __forceinline__ unsigned short f2bf(float f) {
  unsigned u = __float_as_uint(f);
  u += 0x7FFFu + ((u >> 16) & 1u);            // round-to-nearest-even
  return (unsigned short)(u >> 16);
}
__device__ __forceinline__ float bf2f(unsigned short s) {
  return __uint_as_float(((unsigned)s) << 16);
}
__device__ __forceinline__ float dot8_bf(uint4 u, const float* __restrict__ q) {
  float a = bf2f((unsigned short)(u.x & 0xFFFFu)) * q[0];
  a += bf2f((unsigned short)(u.x >> 16))     * q[1];
  a += bf2f((unsigned short)(u.y & 0xFFFFu)) * q[2];
  a += bf2f((unsigned short)(u.y >> 16))     * q[3];
  a += bf2f((unsigned short)(u.z & 0xFFFFu)) * q[4];
  a += bf2f((unsigned short)(u.z >> 16))     * q[5];
  a += bf2f((unsigned short)(u.w & 0xFFFFu)) * q[6];
  a += bf2f((unsigned short)(u.w >> 16))     * q[7];
  return a;
}

// CDNA5 async global->LDS direct copy (16 B per lane), tracked by ASYNCcnt.
// vdst operand = LDS byte address in a VGPR; vaddr = 64-bit global address.
__device__ __forceinline__ void async_copy_b128(unsigned lds_addr, const void* gptr) {
  asm volatile("global_load_async_to_lds_b128 %0, %1, off"
               :: "v"(lds_addr), "v"(gptr) : "memory");
}
__device__ __forceinline__ void wait_async0() {
  asm volatile("s_wait_asynccnt 0" ::: "memory");
}

// ---------------------------------------------------------------------------
// Phase 1: C[M,256] = E[M,256] @ W[256,256] via v_wmma_f32_16x16x32_bf16.
// 64x64 block tile, 256 threads = 8 wave32 waves, 2 16x16 subtiles per wave.
// A staged row-major bf16 in LDS; B staged TRANSPOSED ([n][k]) so each lane's
// 16 K-values form two contiguous runs of 8 k (two ds_load_b128 per fragment half).
// Exactly one of outb / outf is non-null (bf16 vs f32 result).
// ---------------------------------------------------------------------------
__global__ __launch_bounds__(256)
void wmma_gemm_256(const float* __restrict__ E, const float* __restrict__ W,
                   unsigned short* __restrict__ outb, float* __restrict__ outf)
{
  __shared__ unsigned short As[64][40];   // [m][k], padded rows (80 B, 16B-aligned)
  __shared__ unsigned short Bs[64][40];   // [n][k] (W transposed)

  const int m0 = blockIdx.x * 64;
  const int n0 = blockIdx.y * 64;
  const int t  = threadIdx.x;
  const int wv = t >> 5;
  const int lane = t & 31;

  const int s0  = wv * 2;
  const int rs0 = s0 >> 2,       cs0 = s0 & 3;
  const int rs1 = (s0 + 1) >> 2, cs1 = (s0 + 1) & 3;

  v8f acc0 = {0.f,0.f,0.f,0.f,0.f,0.f,0.f,0.f};
  v8f acc1 = {0.f,0.f,0.f,0.f,0.f,0.f,0.f,0.f};

  // ISA 7.12.2 16-bit A layout: lane L holds row (L&15); lanes 0-15 carry
  // K = {0..7,16..23}, lanes 16-31 carry K = {8..15,24..31}.
  const int arow0 = rs0 * 16 + (lane & 15);
  const int arow1 = rs1 * 16 + (lane & 15);
  const int akb   = (lane >> 4) * 8;
  // B layout: lane L holds column (L&15); lanes 0-15 K={0..7,16..23}, 16-31 K={8..15,24..31}.
  const int bcol0 = cs0 * 16 + (lane & 15);
  const int bcol1 = cs1 * 16 + (lane & 15);
  const int bkb   = (lane >> 4) * 8;

  for (int it = 0; it < 8; ++it) {               // K = 256 in steps of 32
    const int k0 = it * 32;
    __syncthreads();
    { // stage A: 64x32 fp32 -> bf16, k-contiguous
      const int row = t >> 2;
      const int ks  = (t & 3) * 8;
      const float4* src = (const float4*)(E + (size_t)(m0 + row) * kD + k0 + ks);
      float4 f0 = src[0], f1 = src[1];
      unsigned short* d = &As[row][ks];
      d[0]=f2bf(f0.x); d[1]=f2bf(f0.y); d[2]=f2bf(f0.z); d[3]=f2bf(f0.w);
      d[4]=f2bf(f1.x); d[5]=f2bf(f1.y); d[6]=f2bf(f1.z); d[7]=f2bf(f1.w);
    }
    { // stage B transposed: W[k][n] -> Bs[n][k]
      const int kk = t >> 3;
      const int ns = (t & 7) * 8;
      const float4* src = (const float4*)(W + (size_t)(k0 + kk) * kD + n0 + ns);
      float4 f0 = src[0], f1 = src[1];
      Bs[ns+0][kk]=f2bf(f0.x); Bs[ns+1][kk]=f2bf(f0.y);
      Bs[ns+2][kk]=f2bf(f0.z); Bs[ns+3][kk]=f2bf(f0.w);
      Bs[ns+4][kk]=f2bf(f1.x); Bs[ns+5][kk]=f2bf(f1.y);
      Bs[ns+6][kk]=f2bf(f1.z); Bs[ns+7][kk]=f2bf(f1.w);
    }
    __syncthreads();

    FragU fa0, fb0, fa1, fb1;
    fa0.q[0] = *(const uint4*)&As[arow0][akb];
    fa0.q[1] = *(const uint4*)&As[arow0][akb + 16];
    fb0.q[0] = *(const uint4*)&Bs[bcol0][bkb];
    fb0.q[1] = *(const uint4*)&Bs[bcol0][bkb + 16];
    fa1.q[0] = *(const uint4*)&As[arow1][akb];
    fa1.q[1] = *(const uint4*)&As[arow1][akb + 16];
    fb1.q[0] = *(const uint4*)&Bs[bcol1][bkb];
    fb1.q[1] = *(const uint4*)&Bs[bcol1][bkb + 16];

    acc0 = __builtin_amdgcn_wmma_f32_16x16x32_bf16(false, fa0.v, false, fb0.v,
                                                   (short)0, acc0, false, false);
    acc1 = __builtin_amdgcn_wmma_f32_16x16x32_bf16(false, fa1.v, false, fb1.v,
                                                   (short)0, acc1, false, false);
  }

  // C/D layout: VGPR v, lane l -> row = v + 8*(l>>4), col = l&15.
  const int rofs = 8 * (lane >> 4);
  const int col  = lane & 15;
  if (outb) {
    #pragma unroll
    for (int v = 0; v < 8; ++v) {
      outb[(size_t)(m0 + rs0*16 + v + rofs) * kD + (n0 + cs0*16 + col)] = f2bf(acc0[v]);
      outb[(size_t)(m0 + rs1*16 + v + rofs) * kD + (n0 + cs1*16 + col)] = f2bf(acc1[v]);
    }
  } else {
    #pragma unroll
    for (int v = 0; v < 8; ++v) {
      outf[(size_t)(m0 + rs0*16 + v + rofs) * kD + (n0 + cs0*16 + col)] = acc0[v];
      outf[(size_t)(m0 + rs1*16 + v + rofs) * kD + (n0 + cs1*16 + col)] = acc1[v];
    }
  }
}

// ---------------------------------------------------------------------------
// Phase 2: greedy decode. One workgroup (256 threads = 8 waves) per batch.
// bf16 K/V/Kt (198 KB) live in the 320 KB WGP LDS for the whole 128-step loop.
// Bulk fill uses CDNA5 GLOBAL_LOAD_ASYNC_TO_LDS_B128 (ASYNCcnt-tracked,
// bypasses VGPRs) followed by s_wait_asynccnt 0 + workgroup barrier.
// ---------------------------------------------------------------------------
__global__ __launch_bounds__(256)
void decode_kernel(const float* __restrict__ emb, const float* __restrict__ demands,
                   const float* __restrict__ WqS, const float* __restrict__ Wo,
                   const unsigned short* __restrict__ Kg,
                   const unsigned short* __restrict__ Vg,
                   const unsigned short* __restrict__ Ktg,
                   const float* __restrict__ Qc,
                   float* __restrict__ outLP, float* __restrict__ outPI)
{
  extern __shared__ unsigned char smem[];
  unsigned short* Kl  = (unsigned short*)smem;            // 129*256 bf16
  unsigned short* Vl  = Kl  + kN * kD;
  unsigned short* Ktl = Vl  + kN * kD;
  float* qf     = (float*)(Ktl + kN * kD);                // 256
  float* curv   = qf     + 256;                           // 256
  float* cmpt   = curv   + 256;                           // 8*132
  float* pw     = cmpt   + 8 * 132;                       // 8*132
  float* attnv  = pw     + 8 * 132;                       // 256
  float* glf    = attnv  + 256;                           // 256
  float* logitv = glf    + 256;                           // 132
  float* maskv  = logitv + 132;                           // 132
  float* hsum   = maskv  + 132;                           // 8
  float* sc     = hsum   + 8;                             // [0]=used [1]=max [2]=log(sum)
  int*   si     = (int*)(sc + 4);                         // [0]=prev

  const int b    = blockIdx.x;
  const int t    = threadIdx.x;
  const int wv   = t >> 5;
  const int lane = t & 31;

  { // async bulk load of this batch's K/V/Kt straight into LDS (16 B / lane)
    const int nvec = kN * kD / 8;                         // 4128 chunks per matrix
    const unsigned lK = (unsigned)(uintptr_t)Kl;          // LDS byte addresses
    const unsigned lV = (unsigned)(uintptr_t)Vl;
    const unsigned lT = (unsigned)(uintptr_t)Ktl;
    const unsigned char* gK = (const unsigned char*)(Kg  + (size_t)b * kN * kD);
    const unsigned char* gV = (const unsigned char*)(Vg  + (size_t)b * kN * kD);
    const unsigned char* gT = (const unsigned char*)(Ktg + (size_t)b * kN * kD);
    for (int i = t; i < nvec; i += 256) {
      const unsigned off = (unsigned)i * 16u;
      async_copy_b128(lK + off, gK + off);
      async_copy_b128(lV + off, gV + off);
      async_copy_b128(lT + off, gT + off);
    }
  }
  // overlap: state init + keep decode weights hot in L2 (global_prefetch_b8)
  for (int n = t; n < 132; n += 256) maskv[n] = 0.f;
  if (t == 0) { sc[0] = 0.f; si[0] = 0; }
  if (t < 32) {
    __builtin_prefetch(WqS + (size_t)t * 2056, 0, 1);
    __builtin_prefetch(Wo  + (size_t)t * 2048, 0, 1);
  }
  wait_async0();
  __syncthreads();

  for (int step = 0; step < kT; ++step) {
    // (a) gather previous node embedding
    curv[t] = emb[((size_t)b * kN + si[0]) * kD + t];
    __syncthreads();

    // (b) Q = Qc + [cur, cap-used] @ Wq_step   (thread t owns output dim t)
    {
      float acc = Qc[(size_t)b * kD + t];
      const float cap = kCap - sc[0];
      #pragma unroll 4
      for (int k = 0; k < kD; ++k) acc += curv[k] * WqS[(size_t)k * kD + t];
      acc += cap * WqS[(size_t)kD * kD + t];
      qf[t] = acc;
    }
    __syncthreads();

    // (c) compat[h][n] = (Q_h . K_h[n]) / 32 + mask
    for (int p = t; p < kH * kN; p += 256) {
      const int h = p / kN;
      const int n = p - h * kN;
      const uint4* kp = (const uint4*)(Kl + (size_t)n * kD + h * kHD);
      const float* qh = qf + h * kHD;
      float acc = 0.f;
      #pragma unroll
      for (int j = 0; j < 4; ++j) acc += dot8_bf(kp[j], qh + j * 8);
      cmpt[h * 132 + n] = acc * (1.0f / kHD) + maskv[n];
    }
    __syncthreads();

    // (d) per-head softmax (wave h owns head h; wave32 shfl reductions)
    {
      float m = -__builtin_inff();
      for (int n = lane; n < kN; n += 32) m = fmaxf(m, cmpt[wv * 132 + n]);
      for (int off = 16; off; off >>= 1) m = fmaxf(m, __shfl_xor(m, off, 32));
      float s = 0.f;
      for (int n = lane; n < kN; n += 32) {
        float e = __expf(cmpt[wv * 132 + n] - m);
        pw[wv * 132 + n] = e;
        s += e;
      }
      for (int off = 16; off; off >>= 1) s += __shfl_xor(s, off, 32);
      if (lane == 0) hsum[wv] = s;
    }
    __syncthreads();

    // (e) attn[h][i] = sum_n p[n] * V_h[n][i]   (thread = (h=wv, i=lane))
    {
      float a = 0.f;
      const unsigned short* vp = Vl + wv * kHD + lane;
      const float* pwh = pw + wv * 132;
      for (int n = 0; n < kN; ++n) a += pwh[n] * bf2f(vp[(size_t)n * kD]);
      attnv[t] = a / hsum[wv];
    }
    __syncthreads();

    // (f) glimpse = attn @ Wo
    {
      float g = 0.f;
      #pragma unroll 4
      for (int k = 0; k < kD; ++k) g += attnv[k] * Wo[(size_t)k * kD + t];
      glf[t] = g;
    }
    __syncthreads();

    // (g) pointer logits: tanh(glimpse . Kt[n] / 16) * 10 + mask
    if (t < kN) {
      const uint4* kp = (const uint4*)(Ktl + (size_t)t * kD);
      float dot = 0.f;
      #pragma unroll 8
      for (int j = 0; j < kD / 8; ++j) dot += dot8_bf(kp[j], glf + j * 8);
      logitv[t] = tanhf(dot * 0.0625f) * kClip + maskv[t];
    }
    __syncthreads();

    // (h) argmax (first-max tie-break) + logsumexp + state update (wave 0)
    if (wv == 0) {
      float m = -__builtin_inff(); int mi = 0;
      for (int n = lane; n < kN; n += 32) {
        float v = logitv[n];
        if (v > m || (v == m && n < mi)) { m = v; mi = n; }
      }
      for (int off = 16; off; off >>= 1) {
        float om = __shfl_xor(m, off, 32);
        int   oi = __shfl_xor(mi, off, 32);
        if (om > m || (om == m && oi < mi)) { m = om; mi = oi; }
      }
      float s = 0.f;
      for (int n = lane; n < kN; n += 32) s += __expf(logitv[n] - m);
      for (int off = 16; off; off >>= 1) s += __shfl_xor(s, off, 32);
      if (lane == 0) {
        const int sel = mi;
        sc[1] = m;
        sc[2] = __logf(s);
        const float dem = demands[(size_t)b * kN + sel];
        sc[0] = (sel == 0) ? 0.f : (sc[0] + dem);
        if (sel != 0) maskv[sel] = -__builtin_inff();
        si[0] = sel;
        outPI[(size_t)b * kT + step] = (float)sel;
      }
    }
    __syncthreads();

    // (i) log_p[b][step][n]
    if (t < kN)
      outLP[((size_t)b * kT + step) * kN + t] = logitv[t] - sc[1] - sc[2];
    __syncthreads();
  }
}

// ---------------------------------------------------------------------------
extern "C" void kernel_launch(void* const* d_in, const int* in_sizes, int n_in,
                              void* d_out, int out_size, void* d_ws, size_t ws_size,
                              hipStream_t stream) {
  const float* emb  = (const float*)d_in[0];   // [B,N,D]
  const float* ctxv = (const float*)d_in[1];   // [B,D]
  const float* dem  = (const float*)d_in[2];   // [B,N]
  const float* WqC  = (const float*)d_in[3];   // [D,D]
  const float* WqS  = (const float*)d_in[4];   // [D+1,D]
  const float* Wk   = (const float*)d_in[5];   // [D,D]
  const float* Wkt  = (const float*)d_in[6];   // [D,D]
  const float* Wv   = (const float*)d_in[7];   // [D,D]
  const float* Wo   = (const float*)d_in[8];   // [D,D]
  float* out = (float*)d_out;

  // workspace: bf16 K/V/Kt (33.8 MB each) + f32 Qc
  unsigned short* Kbf  = (unsigned short*)d_ws;
  unsigned short* Vbf  = Kbf  + (size_t)kB * kN * kD;
  unsigned short* Ktbf = Vbf  + (size_t)kB * kN * kD;
  float*          Qc   = (float*)(Ktbf + (size_t)kB * kN * kD);

  const dim3 blk(256);
  const dim3 gKV(kB * kN / 64, kD / 64);       // 1032 x 4
  hipLaunchKernelGGL(wmma_gemm_256, gKV, blk, 0, stream, emb,  Wk,  Kbf,  (float*)nullptr);
  hipLaunchKernelGGL(wmma_gemm_256, gKV, blk, 0, stream, emb,  Wv,  Vbf,  (float*)nullptr);
  hipLaunchKernelGGL(wmma_gemm_256, gKV, blk, 0, stream, emb,  Wkt, Ktbf, (float*)nullptr);
  const dim3 gQc(kB / 64, kD / 64);            // 8 x 4
  hipLaunchKernelGGL(wmma_gemm_256, gQc, blk, 0, stream, ctxv, WqC, (unsigned short*)nullptr, Qc);

  // dynamic LDS: 3*66048 B bf16 tiles + 13664 B f32 scratch = 211,808 B (< 320 KB/WGP)
  const size_t smem = (size_t)3 * kN * kD * 2 + (size_t)3416 * 4;
  hipFuncSetAttribute(reinterpret_cast<const void*>(decode_kernel),
                      hipFuncAttributeMaxDynamicSharedMemorySize, (int)smem);
  hipLaunchKernelGGL(decode_kernel, dim3(kB), blk, smem, stream,
                     emb, dem, WqS, Wo, Kbf, Vbf, Ktbf, Qc,
                     out, out + (size_t)kB * kT * kN);
}
